// TransformerTower_171798692292
// MI455X (gfx1250) — matrix operand order repaired and checked by reading, not compiled, not run on hardware
//
#include <hip/hip_runtime.h>
#include <hip/hip_bf16.h>

// ---------------------------------------------------------------------------
// TransformerTower for MI455X (gfx1250, wave32, WMMA + async global->LDS).
//
// Algebraic simplification: in _mha, softmax(logits).sum(-1) == 1, so the MHA
// output is one broadcast row derived from sum_s(rms_norm(x)) @ v_w.
// q/k/rope/attn_bias/bias.w are mathematically dead and skipped.
//
// GEMMs: A f32 (converted to bf16 while staging), B pre-materialized bf16
// [N][K] row-major.  B tiles move via GLOBAL_LOAD_ASYNC_TO_LDS_B128 (ASYNCcnt)
// into double-buffered LDS: one barrier per K-step, next tile's async copy and
// A-stage overlap the current tile's WMMA compute.
// ---------------------------------------------------------------------------

#define DIMX 768
#define SEQX 1536

typedef __bf16        bf16x16 __attribute__((ext_vector_type(16)));
typedef float         f32x8   __attribute__((ext_vector_type(8)));
typedef float         f32x4   __attribute__((ext_vector_type(4)));
typedef unsigned int  u32x4   __attribute__((ext_vector_type(4)));

__device__ __forceinline__ unsigned short f2bf(float f) {
  unsigned int u = __float_as_uint(f);
  u += 0x7FFFu + ((u >> 16) & 1u);        // round-to-nearest-even
  return (unsigned short)(u >> 16);
}
__device__ __forceinline__ unsigned pk2(float lo, float hi) {
  return (unsigned)f2bf(lo) | ((unsigned)f2bf(hi) << 16);
}

// ---------------------------------------------------------------------------
// Batched WMMA GEMM: C = epilogue(A[MxK]_f32 @ Bt[NxK]_bf16^T ...)
//   Block: 256 threads (8 waves), tile 64(M) x 128(N), K step 32,
//   double-buffered LDS, one barrier per K-step.
//   BFMODE: 0 = f32 C only
//           1 = f32 C + bf16 row-major mirror Cbf[m*ldbf+n]
//           2 = bf16 row-major only
//           3 = bf16 transposed only  Cbf[n*ldbf+m]   (per-batch)
// Requirements (all call sites): M%64==0, N%128==0, K%32==0.
// ---------------------------------------------------------------------------
template<int ACT, bool HAS_BIAS, bool HAS_ADD, bool HAS_SCALE, int BFMODE>
__global__ __launch_bounds__(256)
void wmma_gemm_kernel(const float* __restrict__ Ab, const unsigned short* __restrict__ Btb,
                      const float* __restrict__ bias, const float* __restrict__ addp,
                      float* __restrict__ Cb, unsigned short* __restrict__ Cbfb,
                      int K, int lda, int ldb, int ldc, int ldbf,
                      long long strA, long long strB, long long strC,
                      long long strAdd, long long strBf, float alpha)
{
  // Double-buffered LDS tiles; row stride 40 elements (80B): 16B aligned rows.
  __shared__ __align__(16) unsigned short sA[2][64][40];   // [m][k] bf16
  __shared__ __align__(16) unsigned short sB[2][128][40];  // [n][k] bf16

  const int tid  = threadIdx.x;
  const int lane = tid & 31;
  const int wave = tid >> 5;
  const int wr   = wave >> 2;          // 0..1
  const int wc   = wave & 3;           // 0..3
  const int bm   = blockIdx.x * 64;
  const int bn   = blockIdx.y * 128;
  const long long bz = blockIdx.z;

  const float*          A  = Ab  + bz * strA;
  const unsigned short* Bt = Btb + bz * strB;
  float*                C  = (BFMODE <= 1) ? (Cb + bz * strC) : (float*)0;
  unsigned short*       CB = (BFMODE >= 1) ? (Cbfb + bz * strBf) : (unsigned short*)0;
  const float*          R  = HAS_ADD ? (addp + bz * strAdd) : (const float*)0;

  f32x8 acc[2][2] = {};

  const int arow = tid >> 2;           // 0..63
  const int acol = (tid & 3) << 3;     // 0,8,16,24
  const int brow = tid >> 1;           // 0..127
  const int bcol = (tid & 1) << 4;     // 0 or 16
  const unsigned long long gbase = (unsigned long long)(uintptr_t)Bt;
  const unsigned loffB = (unsigned)(uintptr_t)&sB[0][brow][bcol];
  const unsigned bufStride = 128u * 40u * 2u;   // bytes between sB buffers

  // Stage tile `it` into LDS buffer `buf`: B via async copy, A via pack+ds_store.
  auto stage = [&](int it, int buf) {
    const int k0 = it << 5;
    const unsigned goff = (unsigned)(((unsigned)(bn + brow) * (unsigned)ldb
                                      + (unsigned)(k0 + bcol)) * 2u);
    const unsigned lo = loffB + (unsigned)buf * bufStride;
    asm volatile(
      "global_load_async_to_lds_b128 %0, %2, %4\n\t"
      "global_load_async_to_lds_b128 %1, %3, %4"
      :: "v"(lo), "v"(lo + 16u), "v"(goff), "v"(goff + 16u), "s"(gbase)
      : "memory");
    const float* pa = A + (long long)(bm + arow) * lda + (k0 + acol);
    f32x4 a0 = *(const f32x4*)pa;
    f32x4 a1 = *(const f32x4*)(pa + 4);
    u32x4 w;
    w[0] = pk2(a0[0], a0[1]); w[1] = pk2(a0[2], a0[3]);
    w[2] = pk2(a1[0], a1[1]); w[3] = pk2(a1[2], a1[3]);
    *(u32x4*)&sA[buf][arow][acol] = w;
  };

  const int nIter = K >> 5;
  stage(0, 0);                                   // prologue

  const int r  = lane & 15;
  const int kh = lane >> 4;
  for (int it = 0; it < nIter; ++it) {
    asm volatile("s_wait_asynccnt 0x0" ::: "memory");   // buf `cur` B resident
    __syncthreads();                                     // + all waves past prior reads
    const int cur = it & 1;
    if (it + 1 < nIter) stage(it + 1, cur ^ 1);          // overlaps compute below

    // ---- fragments per CDNA5 layout ----
    // A 16x32: lanes 0-15 -> M=lane, K 0-7 & 16-23; lanes 16-31 -> K 8-15 & 24-31.
    // B 32x16: lanes 0-15 -> N=lane, K 0-7 & 16-23; lanes 16-31 -> K 8-15 & 24-31.
    bf16x16 af[2], bfr[2];
    #pragma unroll
    for (int mi = 0; mi < 2; ++mi) {
      const unsigned short* p = &sA[cur][wr * 32 + mi * 16 + r][kh << 3];
      union { bf16x16 v; u32x4 u[2]; } t;
      t.u[0] = *(const u32x4*)p;          // K = kh*8 + 0..7
      t.u[1] = *(const u32x4*)(p + 16);   // K = kh*8 + 16..23
      af[mi] = t.v;
    }
    #pragma unroll
    for (int ni = 0; ni < 2; ++ni) {
      const unsigned short* p = &sB[cur][wc * 32 + ni * 16 + r][kh << 3];
      union { bf16x16 v; u32x4 u[2]; } t;
      t.u[0] = *(const u32x4*)p;          // K = kh*8 + 0..7
      t.u[1] = *(const u32x4*)(p + 16);   // K = kh*8 + 16..23
      bfr[ni] = t.v;
    }
    #pragma unroll
    for (int mi = 0; mi < 2; ++mi)
      #pragma unroll
      for (int ni = 0; ni < 2; ++ni)
        acc[mi][ni] = __builtin_amdgcn_wmma_f32_16x16x32_bf16(
            false, af[mi], false, bfr[ni], (short)0, acc[mi][ni], false, false);
  }

  // ---- epilogue: C layout VGPR e -> M = 8*(lane/16)+e, N = lane%16 ----
  // One row base per (mi,e); second ni tile via +16 element immediate offset.
  const int cn  = lane & 15;
  const int m0  = (lane >> 4) << 3;
  const int gn0 = bn + wc * 32 + cn;
  #pragma unroll
  for (int mi = 0; mi < 2; ++mi)
    #pragma unroll
    for (int e = 0; e < 8; ++e) {
      const int gm = bm + wr * 32 + mi * 16 + m0 + e;
      float v0 = acc[mi][0][e];
      float v1 = acc[mi][1][e];
      if (HAS_SCALE) { v0 *= alpha; v1 *= alpha; }
      if (HAS_BIAS)  { v0 += bias[gn0]; v1 += bias[gn0 + 16]; }
      if (ACT == 1)  { v0 = fmaxf(v0, 0.0f); v1 = fmaxf(v1, 0.0f); }
      if (BFMODE <= 1) {
        const long long off = (long long)gm * ldc + gn0;
        if (HAS_ADD) { v0 += R[off]; v1 += R[off + 16]; }
        float* cp = C + off;
        cp[0] = v0; cp[16] = v1;
      }
      if (BFMODE == 1 || BFMODE == 2) {
        unsigned short* bp = CB + (long long)gm * ldbf + gn0;
        bp[0] = f2bf(v0); bp[16] = f2bf(v1);
      }
      if (BFMODE == 3) {
        CB[(long long)gn0 * ldbf + gm]        = f2bf(v0);
        CB[(long long)(gn0 + 16) * ldbf + gm] = f2bf(v1);
      }
    }
}

template<int ACT, bool BI, bool AD, bool SC, int BF>
static inline void gemm(hipStream_t s, const float* A, const unsigned short* Bt,
                        const float* bias, const float* add, float* C, unsigned short* Cbf,
                        int M, int N, int K, int lda, int ldb, int ldc, int ldbf = 0,
                        long long sA = 0, long long sB = 0, long long sC = 0,
                        long long sAdd = 0, long long sBf = 0,
                        int batch = 1, float alpha = 1.0f) {
  dim3 g(M / 64, N / 128, batch);
  wmma_gemm_kernel<ACT, BI, AD, SC, BF><<<g, dim3(256), 0, s>>>(
      A, Bt, bias, add, C, Cbf, K, lda, ldb, ldc, ldbf, sA, sB, sC, sAdd, sBf, alpha);
}

// ---------------------------------------------------------------------------
// Elementwise / reduction kernels
// ---------------------------------------------------------------------------
__global__ void k_fill0(float* p, long long n) {
  long long i = (long long)blockIdx.x * 256 + threadIdx.x;
  if (i < n) p[i] = 0.0f;
}

// bf16 transposed weight: out[n*K+k] = bf16(in[k*N+n])
__global__ void k_wT(const float* __restrict__ in, unsigned short* __restrict__ out,
                     int K, int N) {
  const int i = blockIdx.x * 256 + threadIdx.x;
  if (i >= N * K) return;
  const int n = i / K, k = i - n * K;
  out[i] = f2bf(in[(long long)k * N + n]);
}

__global__ void k_rmsnorm(const float* __restrict__ x, const float* __restrict__ w,
                          float* __restrict__ o, int ncols) {
  __shared__ float red[256];
  const int row = blockIdx.x, t = threadIdx.x;
  const float* xr = x + (long long)row * ncols;
  float s = 0.f;
  for (int c = t; c < ncols; c += 256) { float v = xr[c]; s += v * v; }
  red[t] = s; __syncthreads();
  for (int off = 128; off > 0; off >>= 1) { if (t < off) red[t] += red[t + off]; __syncthreads(); }
  const float inv = rsqrtf(red[0] / (float)ncols + 1e-8f);
  float* orow = o + (long long)row * ncols;
  for (int c = t; c < ncols; c += 256) orow[c] = xr[c] * w[c] * inv;
}

__global__ void k_rowinv(const float* __restrict__ x, float* __restrict__ ri, int ncols) {
  __shared__ float red[256];
  const int row = blockIdx.x, t = threadIdx.x;
  const float* xr = x + (long long)row * ncols;
  float s = 0.f;
  for (int c = t; c < ncols; c += 256) { float v = xr[c]; s += v * v; }
  red[t] = s; __syncthreads();
  for (int off = 128; off > 0; off >>= 1) { if (t < off) red[t] += red[t + off]; __syncthreads(); }
  if (t == 0) ri[row] = rsqrtf(red[0] / (float)ncols + 1e-8f);
}

__global__ void k_colsum(const float* __restrict__ x, const float* __restrict__ ri,
                         const float* __restrict__ nw, float* __restrict__ cs,
                         int nrows, int ncols) {
  const int d = blockIdx.x * 256 + threadIdx.x;
  if (d >= ncols) return;
  float s = 0.f;
  for (int i = 0; i < nrows; ++i) s += x[(long long)i * ncols + d] * ri[i];
  cs[d] = s * nw[d];
}

__global__ void k_vsum(const float* __restrict__ cs, const float* __restrict__ vw,
                       const float* __restrict__ vb, float* __restrict__ vs) {
  const int t = threadIdx.x;
  if (t >= 192) return;
  float s = 0.f;
  for (int k = 0; k < DIMX; ++k) s += cs[k] * vw[k * 192 + t];
  vs[t] = s + (float)SEQX * vb[t];
}

__global__ void k_outrow(const float* __restrict__ vs, const float* __restrict__ ow,
                         const float* __restrict__ ob, float* __restrict__ orow) {
  const int d = blockIdx.x * 256 + threadIdx.x;
  if (d >= DIMX) return;
  float s = ob[d];
  for (int h = 0; h < 8; ++h)
    for (int t = 0; t < 192; ++t)
      s += vs[t] * ow[(long long)(h * 192 + t) * DIMX + d];
  orow[d] = s;
}

__global__ void k_addrow(float* __restrict__ x, const float* __restrict__ row) {
  const long long i = (long long)blockIdx.x * 256 + threadIdx.x;
  if (i < (long long)SEQX * DIMX) x[i] += row[i % DIMX];
}

__global__ void k_downsample(const float* __restrict__ x, float* __restrict__ o) {
  const int i = blockIdx.x * 256 + threadIdx.x;  // < 256*768
  const int p = i / DIMX, d = i - p * DIMX;
  const float* px = x + (long long)p * 6 * DIMX + d;
  float s = 0.f;
  #pragma unroll
  for (int j = 0; j < 6; ++j) s += px[j * DIMX];
  o[i] = s * (1.0f / 6.0f);
}

__global__ void k_gelu(const float* __restrict__ a, float* __restrict__ o, int n) {
  const int i = blockIdx.x * 256 + threadIdx.x;
  if (i >= n) return;
  const float v = a[i];
  o[i] = 0.5f * v * (1.0f + erff(v * 0.70710678118654752f));
}

__global__ void k_addbias(const float* __restrict__ a, const float* __restrict__ b,
                          float* __restrict__ o, int n, int mask) {
  const int i = blockIdx.x * 256 + threadIdx.x;
  if (i < n) o[i] = a[i] + b[i & mask];
}

// pos_enc (bf16, [511][1024]): central_mask_features @ pos_w + pos_b
__global__ void k_posenc(const float* __restrict__ pw, const float* __restrict__ pb,
                         unsigned short* __restrict__ pe) {
  __shared__ float prow[64];
  const int r = blockIdx.x, t = threadIdx.x;
  if (t < 64) {
    const float rel = (float)(r - 255);
    const int j = t & 31;
    const float width = (float)j + powf(225.0f, (float)j * (1.0f / 31.0f));
    const float emb = (width > rel) ? 1.0f : 0.0f;
    const float sgn = (rel > 0.f) ? 1.f : (rel < 0.f ? -1.f : 0.f);
    prow[t] = (t < 32) ? emb : sgn * emb;
  }
  __syncthreads();
  for (int c = t; c < 1024; c += 256) {
    float s = pb[c];
    #pragma unroll 8
    for (int j = 0; j < 64; ++j) s += prow[j] * pw[j * 1024 + c];
    pe[(long long)r * 1024 + c] = f2bf(s);
  }
}

// pair[q,k,:] += (dot + 0.5*(rel_q + rel_k^T)) @ out_w + out_b + yq[q] + yk[k]
__global__ void k_s2p_combine(const float* __restrict__ dots, const float* __restrict__ rq,
                              const float* __restrict__ rk, const float* __restrict__ ow,
                              const float* __restrict__ ob, const float* __restrict__ yq,
                              const float* __restrict__ yk, float* __restrict__ pair) {
  __shared__ float a8[8];
  const int qk = blockIdx.x;            // q*256 + k
  const int q = qk >> 8, k = qk & 255;
  const int t = threadIdx.x;            // 0..127
  if (t < 8) {
    const long long o1 = (long long)t * 65536 + qk;
    const long long o2 = (long long)t * 65536 + k * 256 + q;
    a8[t] = dots[o1] + 0.5f * (rq[o1] + rk[o2]);
  }
  __syncthreads();
  float s = ob[t] + yq[q * 128 + t] + yk[k * 128 + t];
  #pragma unroll
  for (int h = 0; h < 8; ++h) s += a8[h] * ow[h * 128 + t];
  pair[(long long)qk * 128 + t] += s;
}

// row softmax over 256 elements, in place
__global__ void k_softmax256(float* __restrict__ d) {
  __shared__ float red[256];
  const long long row = blockIdx.x;
  const int t = threadIdx.x;
  const float v = d[row * 256 + t];
  red[t] = v; __syncthreads();
  for (int off = 128; off > 0; off >>= 1) { if (t < off) red[t] = fmaxf(red[t], red[t + off]); __syncthreads(); }
  const float m = red[0]; __syncthreads();
  const float e = expf(v - m);
  red[t] = e; __syncthreads();
  for (int off = 128; off > 0; off >>= 1) { if (t < off) red[t] += red[t + off]; __syncthreads(); }
  d[row * 256 + t] = e / red[0];
}

// ---------------------------------------------------------------------------
// Orchestration
// ---------------------------------------------------------------------------
extern "C" void kernel_launch(void* const* d_in, const int* in_sizes, int n_in,
                              void* d_out, int out_size, void* d_ws, size_t ws_size,
                              hipStream_t stream) {
  (void)in_sizes; (void)n_in; (void)out_size; (void)ws_size;

  const float* P_x       = (const float*)d_in[0];
  const float* s2p_qw    = (const float*)d_in[1];
  const float* s2p_kw    = (const float*)d_in[2];
  const float* s2p_posw  = (const float*)d_in[3];
  const float* s2p_posb  = (const float*)d_in[4];
  const float* s2p_qbias = (const float*)d_in[5];
  const float* s2p_kbias = (const float*)d_in[6];
  const float* s2p_outw  = (const float*)d_in[7];
  const float* s2p_outb  = (const float*)d_in[8];
  const float* s2p_yqw   = (const float*)d_in[9];
  const float* s2p_ykw   = (const float*)d_in[10];
  const float* row_qw    = (const float*)d_in[11];
  const float* row_kw    = (const float*)d_in[12];
  const float* row_vw    = (const float*)d_in[13];
  const float* row_vb    = (const float*)d_in[14];
  const float* pm_w1     = (const float*)d_in[15];
  const float* pm_b1     = (const float*)d_in[16];
  const float* pm_w2     = (const float*)d_in[17];
  const float* pm_b2     = (const float*)d_in[18];
  // d_in[19] = bias.w : dead (softmax rows sum to 1 -> MHA independent of logits)
#define MHA_(i,j) ((const float*)d_in[20 + 7*(i) + (j)])  // norm_w,q_w,k_w,v_w,v_b,out_w,out_b
#define MLP_(i,j) ((const float*)d_in[62 + 5*(i) + (j)])  // norm_w,w1,b1,w2,b2

  float* x    = (float*)d_out;                 // 1536 x 768, updated in place
  float* pair = (float*)d_out + 1179648;       // 256 x 256 x 128, updated in place

  float* W = (float*)d_ws;
  // ---- f32 workspace (float offsets) ----
  const long long OXN  = 0;                    // 1536x768 rms-normed x
  const long long OH   = OXN  + 1179648;       // 1536x1536 mlp hidden
  const long long ORI  = OH   + 2359296;       // 1536 inv-rms
  const long long OCS  = ORI  + 1536;          // 768 colsum
  const long long OVS  = OCS  + 768;           // 192 vsum (pad 256)
  const long long OOR  = OVS  + 256;           // 768 broadcast row
  const long long OXDS = OOR  + 768;           // 256x768 downsampled x
  const long long OGX  = OXDS + 196608;        // 256x768 gelu(x_ds)
  const long long OQ   = OGX  + 196608;        // 256x1024 q
  const long long OKm  = OQ   + 262144;        // 256x1024 k (f32 for +bias)
  const long long OQB  = OKm  + 262144;        // q + q_bias
  const long long OKB  = OQB  + 262144;        // k + k_bias
  const long long OYQ  = OKB  + 262144;        // 256x128
  const long long OYK  = OYQ  + 32768;         // 256x128
  const long long ORQ  = OYK  + 32768;         // 8x256x256 rel_q (head-major)
  const long long ORK  = ORQ  + 524288;        // 8x256x256 rel_k
  const long long ODT  = ORK  + 524288;        // 8x256x256 dot
  const long long ORAQ = ODT  + 524288;        // 256x256x128 row-attn q (f32 A)
  const long long OBIG = ORAQ + 8388608;       // 65536x256 logits / pair-mlp hidden
  const long long F32END = OBIG + 16777216;    // 31,788,288 floats

  // ---- bf16 workspace (ushort offsets, 16B-aligned) ----
  unsigned short* U = (unsigned short*)(W + F32END);
  const long long UQWT  = 0;                   // 1024x768  s2p q_w^T
  const long long UKWT  = UQWT  + 786432;      // 1024x768  s2p k_w^T
  const long long UYQT  = UKWT  + 786432;      // 128x768
  const long long UYKT  = UYQT  + 98304;       // 128x768
  const long long URQT  = UYKT  + 98304;       // 128x128 row q_w^T
  const long long URKT  = URQT  + 16384;
  const long long URVT  = URKT  + 16384;
  const long long UPW1T = URVT  + 16384;       // 256x128
  const long long UPW2T = UPW1T + 32768;       // 128x256
  const long long UPE   = UPW2T + 32768;       // 511x1024 pos_enc bf16
  const long long UKM   = UPE   + 523264;      // 256x1024 k bf16 mirror
  const long long URAK  = UKM   + 262144;      // 65536x128 ra_k bf16
  const long long URAVT = URAK  + 8388608;     // 256 x [128][256] ra_v^T bf16
  const long long UW1T  = URAVT + 8388608;     // 6 x 1536x768
  const long long UW2T  = UW1T  + 7077888;     // 6 x 768x1536

  const float inv_sqrt_pd = 0.08838834764831845f;  // 1/sqrt(128)

  hipMemcpyAsync(x, P_x, sizeof(float) * 1179648, hipMemcpyDeviceToDevice, stream);
  k_fill0<<<32768, 256, 0, stream>>>(pair, 8388608LL);
  k_posenc<<<511, 256, 0, stream>>>(s2p_posw, s2p_posb, U + UPE);

  // ---- one-time bf16 transposed weight materialization ----
  k_wT<<<3072, 256, 0, stream>>>(s2p_qw, U + UQWT, 768, 1024);
  k_wT<<<3072, 256, 0, stream>>>(s2p_kw, U + UKWT, 768, 1024);
  k_wT<<<384,  256, 0, stream>>>(s2p_yqw, U + UYQT, 768, 128);
  k_wT<<<384,  256, 0, stream>>>(s2p_ykw, U + UYKT, 768, 128);
  k_wT<<<64,   256, 0, stream>>>(row_qw, U + URQT, 128, 128);
  k_wT<<<64,   256, 0, stream>>>(row_kw, U + URKT, 128, 128);
  k_wT<<<64,   256, 0, stream>>>(row_vw, U + URVT, 128, 128);
  k_wT<<<128,  256, 0, stream>>>(pm_w1, U + UPW1T, 128, 256);
  k_wT<<<128,  256, 0, stream>>>(pm_w2, U + UPW2T, 256, 128);
  for (int L = 0; L < 6; ++L) {
    k_wT<<<4608, 256, 0, stream>>>(MLP_(L, 1), U + UW1T + L * 1179648LL, 768, 1536);
    k_wT<<<4608, 256, 0, stream>>>(MLP_(L, 3), U + UW2T + L * 1179648LL, 1536, 768);
  }

  for (int L = 0; L < 6; ++L) {
    if ((L & 1) == 0) {
      // ---------------- s2p ----------------
      k_downsample<<<768, 256, 0, stream>>>(x, W + OXDS);
      k_gelu<<<768, 256, 0, stream>>>(W + OXDS, W + OGX, 196608);
      gemm<0,0,0,0,0>(stream, W + OXDS, U + UQWT, 0, 0, W + OQ, 0,
                      256, 1024, 768, 768, 768, 1024);
      gemm<0,0,0,0,1>(stream, W + OXDS, U + UKWT, 0, 0, W + OKm, U + UKM,
                      256, 1024, 768, 768, 768, 1024, 1024);
      k_addbias<<<1024, 256, 0, stream>>>(W + OQ,  s2p_qbias, W + OQB, 262144, 1023);
      k_addbias<<<1024, 256, 0, stream>>>(W + OKm, s2p_kbias, W + OKB, 262144, 1023);
      gemm<0,0,0,0,0>(stream, W + OGX, U + UYQT, 0, 0, W + OYQ, 0,
                      256, 128, 768, 768, 768, 128);
      gemm<0,0,0,0,0>(stream, W + OGX, U + UYKT, 0, 0, W + OYK, 0,
                      256, 128, 768, 768, 768, 128);
      // per-head A@B^T (batch = 8 heads; head offset 128 in q/k/pos_enc rows)
      gemm<0,0,0,0,0>(stream, W + OQ,  U + UKM, 0, 0, W + ODT, 0,
                      256, 256, 128, 1024, 1024, 256, 0, 128, 128, 65536, 0, 0, 8);
      gemm<0,0,0,0,0>(stream, W + OQB, U + UPE, 0, 0, W + ORQ, 0,
                      256, 256, 128, 1024, 1024, 256, 0, 128, 128, 65536, 0, 0, 8);
      gemm<0,0,0,0,0>(stream, W + OKB, U + UPE, 0, 0, W + ORK, 0,
                      256, 256, 128, 1024, 1024, 256, 0, 128, 128, 65536, 0, 0, 8);
      k_s2p_combine<<<65536, 128, 0, stream>>>(W + ODT, W + ORQ, W + ORK,
                                               s2p_outw, s2p_outb, W + OYQ, W + OYK, pair);
      // ---------------- row attention ----------------
      gemm<0,0,0,0,0>(stream, pair, U + URQT, 0, 0, W + ORAQ, 0,
                      65536, 128, 128, 128, 128, 128);
      gemm<0,0,0,0,2>(stream, pair, U + URKT, 0, 0, 0, U + URAK,
                      65536, 128, 128, 128, 128, 0, 128);
      gemm<0,1,0,0,3>(stream, pair, U + URVT, row_vb, 0, 0, U + URAVT,   // v^T per batch
                      256, 128, 128, 128, 128, 0, 256, 32768, 0, 0, 0, 32768, 256);
      gemm<0,0,0,1,0>(stream, W + ORAQ, U + URAK, 0, 0, W + OBIG, 0,
                      256, 256, 128, 128, 128, 256, 0, 32768, 32768, 65536, 0, 0, 256,
                      inv_sqrt_pd);
      k_softmax256<<<65536, 256, 0, stream>>>(W + OBIG);
      gemm<0,0,1,0,0>(stream, W + OBIG, U + URAVT, 0, pair, pair, 0,
                      256, 128, 256, 256, 256, 128, 0, 65536, 32768, 32768, 32768, 0, 256);
      // ---------------- pair MLP ----------------
      gemm<1,1,0,0,0>(stream, pair, U + UPW1T, pm_b1, 0, W + OBIG, 0,
                      65536, 256, 128, 128, 128, 256);
      gemm<0,1,1,0,0>(stream, W + OBIG, U + UPW2T, pm_b2, pair, pair, 0,
                      65536, 128, 256, 256, 256, 128);
    }

    // ---------------- MHA (collapsed: softmax rows sum to 1) ----------------
    k_rowinv<<<1536, 256, 0, stream>>>(x, W + ORI, DIMX);
    k_colsum<<<3, 256, 0, stream>>>(x, W + ORI, MHA_(L, 0), W + OCS, SEQX, DIMX);
    k_vsum  <<<1, 192, 0, stream>>>(W + OCS, MHA_(L, 3), MHA_(L, 4), W + OVS);
    k_outrow<<<3, 256, 0, stream>>>(W + OVS, MHA_(L, 5), MHA_(L, 6), W + OOR);
    k_addrow<<<4608, 256, 0, stream>>>(x, W + OOR);

    // ---------------- token MLP ----------------
    k_rmsnorm<<<1536, 256, 0, stream>>>(x, MLP_(L, 0), W + OXN, DIMX);
    gemm<1,1,0,0,0>(stream, W + OXN, U + UW1T + L * 1179648LL, MLP_(L, 2), 0, W + OH, 0,
                    SEQX, 2 * DIMX, DIMX, DIMX, DIMX, 2 * DIMX);
    gemm<0,1,1,0,0>(stream, W + OH, U + UW2T + L * 1179648LL, MLP_(L, 4), x, x, 0,
                    SEQX, DIMX, 2 * DIMX, 2 * DIMX, 2 * DIMX, DIMX);
  }
#undef MHA_
#undef MLP_
}
